// DLRMResNet_48876727828683
// MI455X (gfx1250) — matrix operand order repaired and checked by reading, not compiled, not run on hardware
//
#include <hip/hip_runtime.h>
#include <hip/hip_bf16.h>

// ---------------------------------------------------------------------------
// DLRM-ResNet forward for MI455X (gfx1250, wave32, WMMA).
// bf16 activations/weights, f32 accumulation via v_wmma_f32_16x16x32_bf16.
// Top-MLP weight tiles staged to LDS with GLOBAL_LOAD_ASYNC_TO_LDS_B128
// (ASYNCcnt) and double-buffered against WMMA compute.
// ---------------------------------------------------------------------------

typedef __attribute__((ext_vector_type(16))) __bf16 bf16x16;
typedef __attribute__((ext_vector_type(8)))  __bf16 bf16x8;
typedef __attribute__((ext_vector_type(4)))  __bf16 bf16x4;
typedef __attribute__((ext_vector_type(8)))  float  f32x8;

#define VOCAB    2097152
#define CHUNK    (VOCAB / 4)        // 524288 = 2^19
#define EMBED    128
#define NDENSE   13
#define NSPARSE  26
#define NFEAT    (NDENSE + NSPARSE) // 39
#define BATCH    32768
#define ZDIM     (256 + NSPARSE * EMBED)  // 3584

// ---- WMMA helpers ----------------------------------------------------------
__device__ __forceinline__ f32x8 wmma_bf16(bf16x16 a, bf16x16 b, f32x8 c) {
  // (neg_a, A, neg_b, B, c_mod, C, reuse_a, reuse_b)
  return __builtin_amdgcn_wmma_f32_16x16x32_bf16(false, a, false, b, (short)0, c,
                                                 false, false);
}

union FragU { bf16x16 v; bf16x8 h[2]; };

// A: 16x32 bf16 tile, row-major with leading dim ld (elements).
// lane layout: row = lane&15, K window = 8*(lane>>4), per ISA 7.12.2.
__device__ __forceinline__ bf16x16 load_a_frag(const __bf16* base, int ld,
                                               int m, int lh) {
  FragU u;
  u.h[0] = *(const bf16x8*)(base + (size_t)m * ld + lh * 8);
  u.h[1] = *(const bf16x8*)(base + (size_t)m * ld + 16 + lh * 8);
  return u.v;
}

// B: 32x16 bf16 tile taken from an N-major (N x K row-major) matrix.
// lane layout: col = lane&15, K window = 8*(lane>>4), per ISA 7.12.2.
__device__ __forceinline__ bf16x16 load_b_frag(const __bf16* base, int ld,
                                               int n, int lh) {
  FragU u;
  u.h[0] = *(const bf16x8*)(base + (size_t)n * ld + lh * 8);
  u.h[1] = *(const bf16x8*)(base + (size_t)n * ld + 16 + lh * 8);
  return u.v;
}

// ---- CDNA5 async Global->LDS copy (CDNA5 ISA) --------
// VDST = LDS byte address (dynamic LDS starts at offset 0 here), GV mode
// (64-bit vaddr, SADDR = off). Tracked by ASYNCcnt.
__device__ __forceinline__ void issue_async_b128(unsigned lds_byte_off,
                                                 const void* g) {
  unsigned long long ga = (unsigned long long)(uintptr_t)g;
  asm volatile("global_load_async_to_lds_b128 %0, %1, off"
               :: "v"(lds_byte_off), "v"(ga) : "memory");
}
__device__ __forceinline__ void wait_async0() {
  asm volatile("s_wait_asynccnt 0x0" ::: "memory");
}

// ---- weight conversion -----------------------------------------------------
__global__ void __launch_bounds__(256)
convert_f32_bf16_kernel(const float* __restrict__ src, __bf16* __restrict__ dst,
                        int n) {
  int i = blockIdx.x * blockDim.x + threadIdx.x;
  int stride = gridDim.x * blockDim.x;
  for (; i < n; i += stride) dst[i] = (__bf16)src[i];
}

// bw0 is (256 x 13); pad K to 32 with zeros for a single WMMA K-step.
__global__ void __launch_bounds__(256)
pad_bw0_kernel(const float* __restrict__ src, __bf16* __restrict__ dst) {
  int i = blockIdx.x * blockDim.x + threadIdx.x;
  if (i >= 256 * 32) return;
  int nrow = i >> 5, k = i & 31;
  dst[i] = (k < NDENSE) ? (__bf16)src[nrow * NDENSE + k] : (__bf16)0.0f;
}

// ---- fused bottom MLP: 13 -> 256 -> 256 -> 256 (ReLU + residual) -----------
// One workgroup = 64 rows, 4 waves, h kept in 64 KB LDS (never hits HBM).
// Output written into z[:, 0:256] (bf16).
__global__ void __launch_bounds__(128)
bottom_mlp_kernel(const float* __restrict__ x,
                  const __bf16* __restrict__ w0p,  // 256 x 32 (padded)
                  const float* __restrict__ bb0,
                  const __bf16* __restrict__ w1,   // 256 x 256
                  const float* __restrict__ bb1,
                  const __bf16* __restrict__ w2,   // 256 x 256
                  const float* __restrict__ bb2,
                  __bf16* __restrict__ z) {
  extern __shared__ char smem[];
  __bf16* hA = (__bf16*)smem;     // 64 x 256 bf16 = 32 KB
  __bf16* hB = hA + 64 * 256;     // 64 x 256 bf16 = 32 KB
  __bf16* dt = hB;                // dense staging (64 x 32) aliases hB

  const int tid  = threadIdx.x;
  const int wave = tid >> 5;
  const int lane = tid & 31;
  const int m    = lane & 15;     // tile row (A) / tile col (B,D)
  const int lh   = lane >> 4;     // lane half
  const int row0 = blockIdx.x * 64;
  const int mrow = wave * 16;     // this wave's 16-row M-tile

  // Stage dense features, pad K: 13 -> 32.
  for (int i = tid; i < 64 * 32; i += 128) {
    int r = i >> 5, k = i & 31;
    float v = (k < NDENSE) ? x[(size_t)(row0 + r) * NFEAT + k] : 0.0f;
    dt[i] = (__bf16)v;
  }
  __syncthreads();

  // ---- layer 0: hA = relu(dense @ w0^T + b0), K = 32 (1 WMMA step) ----
  {
    bf16x16 a = load_a_frag(dt + mrow * 32, 32, m, lh);
    for (int nt = 0; nt < 16; ++nt) {
      bf16x16 b = load_b_frag(w0p, 32, nt * 16 + m, lh);
      f32x8 acc = {};
      acc = wmma_bf16(a, b, acc);
      int col = nt * 16 + m;
      float bias = bb0[col];
#pragma unroll
      for (int r = 0; r < 8; ++r) {
        float v = acc[r] + bias;
        v = v > 0.0f ? v : 0.0f;
        hA[(mrow + r + 8 * lh) * 256 + col] = (__bf16)v;
      }
    }
  }
  __syncthreads();

  // ---- layer 1: hB = relu(hA @ w1^T + b1) + hA, K = 256 ----
  for (int nt = 0; nt < 16; ++nt) {
    f32x8 acc = {};
#pragma unroll
    for (int ks = 0; ks < 8; ++ks) {
      bf16x16 a = load_a_frag(hA + mrow * 256 + ks * 32, 256, m, lh);
      bf16x16 b = load_b_frag(w1 + ks * 32, 256, nt * 16 + m, lh);
      acc = wmma_bf16(a, b, acc);
    }
    int col = nt * 16 + m;
    float bias = bb1[col];
#pragma unroll
    for (int r = 0; r < 8; ++r) {
      int row = (mrow + r + 8 * lh) * 256 + col;
      float v = acc[r] + bias;
      v = v > 0.0f ? v : 0.0f;
      v += (float)hA[row];
      hB[row] = (__bf16)v;
    }
  }
  __syncthreads();

  // ---- layer 2: hA = relu(hB @ w2^T + b2) + hB, K = 256 ----
  for (int nt = 0; nt < 16; ++nt) {
    f32x8 acc = {};
#pragma unroll
    for (int ks = 0; ks < 8; ++ks) {
      bf16x16 a = load_a_frag(hB + mrow * 256 + ks * 32, 256, m, lh);
      bf16x16 b = load_b_frag(w2 + ks * 32, 256, nt * 16 + m, lh);
      acc = wmma_bf16(a, b, acc);
    }
    int col = nt * 16 + m;
    float bias = bb2[col];
#pragma unroll
    for (int r = 0; r < 8; ++r) {
      int row = (mrow + r + 8 * lh) * 256 + col;
      float v = acc[r] + bias;
      v = v > 0.0f ? v : 0.0f;
      v += (float)hB[row];
      hA[row] = (__bf16)v;
    }
  }
  __syncthreads();

  // Copy h into z[:, 0:256] as bf16 (128-bit stores).
  for (int i = tid; i < 64 * 32; i += 128) {
    int r = i >> 5, c8 = (i & 31) * 8;
    *(bf16x8*)(z + (size_t)(row0 + r) * ZDIM + c8) =
        *(const bf16x8*)(hA + r * 256 + c8);
  }
}

// ---- embedding gather: one wave per 512 B embedding row --------------------
__global__ void __launch_bounds__(256)
gather_kernel(const float* __restrict__ x, const float* __restrict__ e0,
              const float* __restrict__ e1, const float* __restrict__ e2,
              const float* __restrict__ e3, __bf16* __restrict__ z) {
  int g    = blockIdx.x * 8 + (threadIdx.x >> 5);  // global gather id
  int lane = threadIdx.x & 31;
  int row  = g / NSPARSE;
  int f    = g - row * NSPARSE;
  int idx  = (int)x[(size_t)row * NFEAT + NDENSE + f];
  idx &= (VOCAB - 1);                // % VOCAB (power of two)
  int c   = idx >> 19;               // chunk
  int off = idx & (CHUNK - 1);
  const float* tab = (c == 0) ? e0 : (c == 1) ? e1 : (c == 2) ? e2 : e3;
  float4 v = *(const float4*)(tab + (size_t)off * EMBED + lane * 4);
  bf16x4 o;
  o[0] = (__bf16)v.x; o[1] = (__bf16)v.y; o[2] = (__bf16)v.z; o[3] = (__bf16)v.w;
  *(bf16x4*)(z + (size_t)row * ZDIM + 256 + f * EMBED + lane * 4) = o;
}

// ---- top MLP GEMM: out = relu(A @ W^T + bias) [+ res], N = 256 -------------
// Block = 8 waves = 128 rows; each wave owns a 16x256 output tile so A (z) is
// streamed from HBM exactly once. The 256x32 weight tile (16 KB) is staged in
// LDS with async Global->LDS b128 copies, double-buffered against WMMA.
__device__ __forceinline__ void issue_w_tile(const __bf16* __restrict__ W,
                                             int K, int ks, unsigned buf_off,
                                             int tid) {
  // 256 cols x 32 K bf16 = 1024 x b128 chunks; 4 per thread.
  for (int i = tid; i < 1024; i += 256) {
    int c = i >> 2, kq = i & 3;
    issue_async_b128(buf_off + (unsigned)(c * 64 + kq * 16),
                     W + (size_t)c * K + ks * 32 + kq * 8);
  }
}

__global__ void __launch_bounds__(256)
top_gemm_kernel(const __bf16* __restrict__ A,    // M x K bf16
                const __bf16* __restrict__ W,    // 256 x K bf16
                const float* __restrict__ bias,  // 256
                const __bf16* __restrict__ res,  // M x 256 or nullptr
                __bf16* __restrict__ out,        // M x 256 bf16
                int K) {
  extern __shared__ char smem[];
  __bf16* ws = (__bf16*)smem;  // 2 x (256 x 32) bf16 = 32 KB, double-buffered

  const int tid  = threadIdx.x;
  const int wave = tid >> 5;
  const int lane = tid & 31;
  const int m    = lane & 15;
  const int lh   = lane >> 4;
  const int row0 = blockIdx.x * 128;
  const int mrow = wave * 16;
  const __bf16* Arow = A + (size_t)(row0 + mrow + m) * K;
  const int nks = K >> 5;

  issue_w_tile(W, K, 0, 0u, tid);  // prologue: stage K-step 0 into buffer 0

  f32x8 acc[16] = {};
  for (int ks = 0; ks < nks; ++ks) {
    wait_async0();      // own async copies for buf[ks&1] have landed
    __syncthreads();    // everyone's copies landed; prev compute finished
    if (ks + 1 < nks)   // overlap next tile's DMA with this tile's WMMAs
      issue_w_tile(W, K, ks + 1, ((ks + 1) & 1) ? 16384u : 0u, tid);

    const __bf16* cur = ws + (ks & 1) * 8192;
    FragU ua;
    ua.h[0] = *(const bf16x8*)(Arow + ks * 32 + lh * 8);
    ua.h[1] = *(const bf16x8*)(Arow + ks * 32 + 16 + lh * 8);
#pragma unroll
    for (int j = 0; j < 16; ++j) {
      bf16x16 b = load_b_frag(cur, 32, j * 16 + m, lh);
      acc[j] = wmma_bf16(ua.v, b, acc[j]);
    }
  }

  // Epilogue: bias + ReLU (+ residual), bf16 store.
#pragma unroll
  for (int j = 0; j < 16; ++j) {
    int col = j * 16 + m;
    float bv = bias[col];
#pragma unroll
    for (int r = 0; r < 8; ++r) {
      int row = row0 + mrow + r + 8 * lh;
      float v = acc[j][r] + bv;
      v = v > 0.0f ? v : 0.0f;
      if (res) v += (float)res[(size_t)row * 256 + col];
      out[(size_t)row * 256 + col] = (__bf16)v;
    }
  }
}

// ---- final layer (N = 1): wave-parallel dot product ------------------------
__global__ void __launch_bounds__(256)
final_dot_kernel(const __bf16* __restrict__ A, const __bf16* __restrict__ w,
                 const float* __restrict__ b, float* __restrict__ out) {
  int gw   = (blockIdx.x * blockDim.x + threadIdx.x) >> 5;  // one wave per row
  int lane = threadIdx.x & 31;
  bf16x8 a  = *(const bf16x8*)(A + (size_t)gw * 256 + lane * 8);
  bf16x8 wv = *(const bf16x8*)(w + lane * 8);
  float s = 0.0f;
#pragma unroll
  for (int i = 0; i < 8; ++i) s += (float)a[i] * (float)wv[i];
#pragma unroll
  for (int off = 16; off > 0; off >>= 1) s += __shfl_xor(s, off, 32);
  if (lane == 0) out[gw] = s + b[0];
}

// ---------------------------------------------------------------------------
extern "C" void kernel_launch(void* const* d_in, const int* in_sizes, int n_in,
                              void* d_out, int out_size, void* d_ws,
                              size_t ws_size, hipStream_t stream) {
  (void)in_sizes; (void)n_in; (void)out_size; (void)ws_size;
  const float* x   = (const float*)d_in[0];
  const float* e0  = (const float*)d_in[1];
  const float* e1  = (const float*)d_in[2];
  const float* e2  = (const float*)d_in[3];
  const float* e3  = (const float*)d_in[4];
  const float* bw0 = (const float*)d_in[5];
  const float* bb0 = (const float*)d_in[6];
  const float* bw1 = (const float*)d_in[7];
  const float* bb1 = (const float*)d_in[8];
  const float* bw2 = (const float*)d_in[9];
  const float* bb2 = (const float*)d_in[10];
  const float* tw0 = (const float*)d_in[11];
  const float* tb0 = (const float*)d_in[12];
  const float* tw1 = (const float*)d_in[13];
  const float* tb1 = (const float*)d_in[14];
  const float* tw2 = (const float*)d_in[15];
  const float* tb2 = (const float*)d_in[16];
  const float* tw3 = (const float*)d_in[17];
  const float* tb3 = (const float*)d_in[18];
  const float* tw4 = (const float*)d_in[19];
  const float* tb4 = (const float*)d_in[20];

  // Workspace layout (bf16 regions), 256-byte aligned.
  char* p = (char*)d_ws;
  auto take = [&](size_t bytes) {
    char* r = p;
    p += (bytes + 255) & ~(size_t)255;
    return r;
  };
  __bf16* z   = (__bf16*)take((size_t)BATCH * ZDIM * 2);  // 224 MB
  __bf16* hA  = (__bf16*)take((size_t)BATCH * 256 * 2);   // 16 MB
  __bf16* hB  = (__bf16*)take((size_t)BATCH * 256 * 2);   // 16 MB
  __bf16* w0p = (__bf16*)take((size_t)256 * 32 * 2);
  __bf16* w1b = (__bf16*)take((size_t)256 * 256 * 2);
  __bf16* w2b = (__bf16*)take((size_t)256 * 256 * 2);
  __bf16* t0b = (__bf16*)take((size_t)256 * ZDIM * 2);    // 1.75 MB (L2-hot)
  __bf16* t1b = (__bf16*)take((size_t)256 * 256 * 2);
  __bf16* t2b = (__bf16*)take((size_t)256 * 256 * 2);
  __bf16* t3b = (__bf16*)take((size_t)256 * 256 * 2);
  __bf16* t4b = (__bf16*)take((size_t)256 * 2);

  // Weight conversion to bf16 (tiny; all weights stay L2-resident).
  pad_bw0_kernel<<<32, 256, 0, stream>>>(bw0, w0p);
  convert_f32_bf16_kernel<<<256, 256, 0, stream>>>(bw1, w1b, 256 * 256);
  convert_f32_bf16_kernel<<<256, 256, 0, stream>>>(bw2, w2b, 256 * 256);
  convert_f32_bf16_kernel<<<1024, 256, 0, stream>>>(tw0, t0b, 256 * ZDIM);
  convert_f32_bf16_kernel<<<256, 256, 0, stream>>>(tw1, t1b, 256 * 256);
  convert_f32_bf16_kernel<<<256, 256, 0, stream>>>(tw2, t2b, 256 * 256);
  convert_f32_bf16_kernel<<<256, 256, 0, stream>>>(tw3, t3b, 256 * 256);
  convert_f32_bf16_kernel<<<1, 256, 0, stream>>>(tw4, t4b, 256);

  // Fused bottom MLP -> z[:, 0:256]   (64 KB dynamic LDS per block)
  bottom_mlp_kernel<<<BATCH / 64, 128, 64 * 1024, stream>>>(
      x, w0p, bb0, w1b, bb1, w2b, bb2, z);

  // Embedding gathers -> z[:, 256:3584]
  gather_kernel<<<(BATCH * NSPARSE) / 8, 256, 0, stream>>>(x, e0, e1, e2, e3, z);

  // Top MLP (32 KB dynamic LDS: double-buffered async weight tiles)
  top_gemm_kernel<<<BATCH / 128, 256, 32768, stream>>>(z,  t0b, tb0, nullptr,
                                                       hA, ZDIM);
  top_gemm_kernel<<<BATCH / 128, 256, 32768, stream>>>(hA, t1b, tb1, hA, hB,
                                                       256);
  top_gemm_kernel<<<BATCH / 128, 256, 32768, stream>>>(hB, t2b, tb2, hB, hA,
                                                       256);
  top_gemm_kernel<<<BATCH / 128, 256, 32768, stream>>>(hA, t3b, tb3, hA, hB,
                                                       256);
  final_dot_kernel<<<BATCH / 8, 256, 0, stream>>>(hB, t4b, tb4, (float*)d_out);
}
